// SelfAttentionBlock2D_9208409883344
// MI455X (gfx1250) — matrix operand order repaired and checked by reading, not compiled, not run on hardware
//
#include <hip/hip_runtime.h>
#include <hip/hip_bf16.h>

typedef __bf16 bf16_t;
typedef __attribute__((ext_vector_type(16))) __bf16 v16bf;
typedef __attribute__((ext_vector_type(8)))  __bf16 v8bf;
typedef __attribute__((ext_vector_type(8)))  float  v8f;

// ---------------------------------------------------------------------------
// WMMA helpers (CDNA5 wave32, V_WMMA_F32_16X16X32_BF16)
// A-fragment per-lane layout (ISA 7.12.2, 16-bit A 16x32): lane L -> row M=L%16,
//   elements 0..7  = K  k0 + 8*(L/16) .. +7
//   elements 8..15 = K  k0 + 16 + 8*(L/16) .. +7
// B-fragment (K x 16, col N=L%16): same K order as A (j<8: k0+8*(L/16)+j, j>=8: k0+16+8*(L/16)+j-8)
// C/D f32: VGPR r, lane L -> row M = r + 8*(L/16), col N = L%16
// ---------------------------------------------------------------------------

__device__ __forceinline__ v8f vzero8() {
  v8f z;
#pragma unroll
  for (int i = 0; i < 8; ++i) z[i] = 0.0f;
  return z;
}

__device__ __forceinline__ v16bf load_fragA(const bf16_t* base, int ld, int k0) {
  const int lane = threadIdx.x & 31;
  const int r = lane & 15, hh = lane >> 4;
  const bf16_t* p = base + (size_t)r * (size_t)ld + (size_t)(k0 + hh * 8);
  union { v16bf v; v8bf h2[2]; } u;
  u.h2[0] = *(const v8bf*)(p);
  u.h2[1] = *(const v8bf*)(p + 16);
  return u.v;
}

__device__ __forceinline__ v16bf load_fragB(const bf16_t* base, int ld, int k0) {
  const int lane = threadIdx.x & 31;
  const int r = lane & 15, hh = lane >> 4;
  const bf16_t* p = base + (size_t)r * (size_t)ld + (size_t)(k0 + hh * 8);
  union { v16bf v; v8bf h2[2]; } u;
  u.h2[0] = *(const v8bf*)(p);
  u.h2[1] = *(const v8bf*)(p + 16);
  return u.v;
}

__device__ __forceinline__ v8f wmma_bf16(v16bf a, v16bf b, v8f c) {
  return __builtin_amdgcn_wmma_f32_16x16x32_bf16(false, a, false, b, (short)0, c,
                                                 false, false);
}

// ---------------------------------------------------------------------------
// Prep kernels: BN folding + bf16 weight conversion
// ---------------------------------------------------------------------------

// P layout (floats): sk1[256] shk1[256] sk2[256] shk2[256]
//                    sq1[256] shq1[256] sq2[256] shq2[256]
//                    wos[512] wosh[512]
__global__ void bnparams_kernel(const float* __restrict__ kb1, const float* __restrict__ kb2,
                                const float* __restrict__ qb1, const float* __restrict__ qb2,
                                const float* __restrict__ wob, float* __restrict__ P) {
  int i = blockIdx.x * blockDim.x + threadIdx.x;
  const float* src; int C, c; float *sout, *shout;
  if (i < 256)       { src = kb1; C = 256; c = i;        sout = P;        shout = P + 256;  }
  else if (i < 512)  { src = kb2; C = 256; c = i - 256;  sout = P + 512;  shout = P + 768;  }
  else if (i < 768)  { src = qb1; C = 256; c = i - 512;  sout = P + 1024; shout = P + 1280; }
  else if (i < 1024) { src = qb2; C = 256; c = i - 768;  sout = P + 1536; shout = P + 1792; }
  else if (i < 1536) { src = wob; C = 512; c = i - 1024; sout = P + 2048; shout = P + 2560; }
  else return;
  float g = src[c], bb = src[C + c], m = src[2 * C + c], v = src[3 * C + c];
  float sc = g * rsqrtf(v + 1e-5f);
  sout[c] = sc;
  shout[c] = bb - m * sc;
}

// A[o,c] = s2[o] * sum_j W2[o,j]*s1[j]*W1[j,c]  (256x512, bf16)
// bias[o] = s2[o] * sum_j W2[o,j]*sh1[j] + sh2[o]
__global__ void fold_kernel(const float* __restrict__ W1, const float* __restrict__ W2,
                            const float* __restrict__ s1, const float* __restrict__ sh1,
                            const float* __restrict__ s2, const float* __restrict__ sh2,
                            bf16_t* __restrict__ A, float* __restrict__ bias) {
  int tid = blockIdx.x * blockDim.x + threadIdx.x;  // 256*512 threads
  int o = tid >> 9, c = tid & 511;
  float acc = 0.f;
  for (int j = 0; j < 256; ++j) acc += W2[o * 256 + j] * s1[j] * W1[j * 512 + c];
  A[o * 512 + c] = (bf16_t)(s2[o] * acc);
  if (c == 0) {
    float bb = 0.f;
    for (int j = 0; j < 256; ++j) bb += W2[o * 256 + j] * sh1[j];
    bias[o] = s2[o] * bb + sh2[o];
  }
}

__global__ void cvt_kernel(const float* __restrict__ src, bf16_t* __restrict__ dst, int n) {
  int i = blockIdx.x * blockDim.x + threadIdx.x;
  if (i < n) dst[i] = (bf16_t)src[i];
}

// ---------------------------------------------------------------------------
// Projection: Qt[n,c]=Aq@x+bq, Kt[n,c]=Ak@x+bk (n-major, c contiguous),
//             V[c,n]=Wv@x (c-major, n contiguous). blockIdx.z selects mode.
// 256 threads = 8 waves; 128 n-columns of x per block; Cin=512 staged as
// transposed bf16 tiles [128 n][64 cin] in LDS.
// ---------------------------------------------------------------------------
__global__ __launch_bounds__(256) void proj_kernel(
    const float* __restrict__ x, const bf16_t* __restrict__ Wq,
    const bf16_t* __restrict__ Wk, const bf16_t* __restrict__ Wv,
    const float* __restrict__ bq, const float* __restrict__ bk,
    bf16_t* __restrict__ Qt, bf16_t* __restrict__ Kt, bf16_t* __restrict__ Vout,
    int N) {
  const int b = blockIdx.y;
  const int n0 = blockIdx.x * 128;
  const int mode = blockIdx.z;  // 0=Q, 1=K, 2=V
  const int w = threadIdx.x >> 5;
  const int lane = threadIdx.x & 31, r16 = lane & 15, hh = lane >> 4;

  __shared__ __align__(16) bf16_t sX[128][72];  // [n][cin], padded

  const bf16_t* W = (mode == 0) ? Wq : ((mode == 1) ? Wk : Wv);
  const float* xb = x + (size_t)b * 512 * N;

  v8f acc[16];
#pragma unroll
  for (int t = 0; t < 16; ++t) acc[t] = vzero8();

  for (int c0 = 0; c0 < 512; c0 += 64) {
    __syncthreads();
#pragma unroll
    for (int i = 0; i < 32; ++i) {
      int idx = (int)threadIdx.x + i * 256;
      int ci = idx >> 7, nj = idx & 127;
      sX[nj][ci] = (bf16_t)xb[(size_t)(c0 + ci) * N + n0 + nj];
    }
    __syncthreads();

    if (mode < 2) {
      // M = n rows (wave's 16), N-dim = 256 output channels, K = cin
#pragma unroll
      for (int kc = 0; kc < 2; ++kc) {
        v16bf a = load_fragA(&sX[w * 16][0], 72, kc * 32);
#pragma unroll
        for (int t = 0; t < 16; ++t) {
          v16bf bw = load_fragB(W + (size_t)(t * 16) * 512 + c0, 512, kc * 32);
          acc[t] = wmma_bf16(a, bw, acc[t]);
        }
      }
    } else {
      // M = output channel c (wave owns 32), N-dim = n (128 cols), K = cin
#pragma unroll
      for (int kc = 0; kc < 2; ++kc) {
#pragma unroll
        for (int mt = 0; mt < 2; ++mt) {
          v16bf a = load_fragA(W + (size_t)(w * 32 + mt * 16) * 512 + c0, 512, kc * 32);
#pragma unroll
          for (int nt = 0; nt < 8; ++nt) {
            v16bf bx = load_fragB(&sX[nt * 16][0], 72, kc * 32);
            acc[mt * 8 + nt] = wmma_bf16(a, bx, acc[mt * 8 + nt]);
          }
        }
      }
    }
  }

  if (mode < 2) {
    const float* bias = (mode == 0) ? bq : bk;
    bf16_t* Out = ((mode == 0) ? Qt : Kt) + (size_t)b * N * 256 +
                  (size_t)(n0 + w * 16) * 256;
#pragma unroll
    for (int t = 0; t < 16; ++t) {
      float bv = bias[t * 16 + r16];
#pragma unroll
      for (int r = 0; r < 8; ++r)
        Out[(size_t)(r + 8 * hh) * 256 + t * 16 + r16] = (bf16_t)(acc[t][r] + bv);
    }
  } else {
    bf16_t* Out = Vout + (size_t)b * 256 * N;
#pragma unroll
    for (int mt = 0; mt < 2; ++mt)
#pragma unroll
      for (int nt = 0; nt < 8; ++nt)
#pragma unroll
        for (int r = 0; r < 8; ++r)
          Out[(size_t)(w * 32 + mt * 16 + r + 8 * hh) * N + n0 + nt * 16 + r16] =
              (bf16_t)acc[mt * 8 + nt][r];
  }
}

// ---------------------------------------------------------------------------
// Flash attention: each wave owns 16 query rows; streams 64-key blocks from
// L2-resident Kt/V; online softmax; P round-trips through per-wave LDS tile.
// Ct[n, v] = softmax(Qt@Kt^T / 16) @ V^T   (bf16 out, n-major)
// ---------------------------------------------------------------------------
#define ATT_BN 64
__global__ __launch_bounds__(256) void attn_kernel(
    const bf16_t* __restrict__ Qt, const bf16_t* __restrict__ Kt,
    const bf16_t* __restrict__ Vv, bf16_t* __restrict__ Ct, int N) {
  const int b = blockIdx.y;
  const int q0 = blockIdx.x * 128;
  const int w = threadIdx.x >> 5;
  const int lane = threadIdx.x & 31, r16 = lane & 15, hh = lane >> 4;

  __shared__ __align__(16) bf16_t sP[8][16][ATT_BN];  // per-wave P tile

  const bf16_t* Qw = Qt + (size_t)b * N * 256 + (size_t)(q0 + w * 16) * 256;
  const bf16_t* Kb = Kt + (size_t)b * N * 256;
  const bf16_t* Vb = Vv + (size_t)b * 256 * N;

  v8f o[16];
#pragma unroll
  for (int t = 0; t < 16; ++t) o[t] = vzero8();
  float mrow[8], lrow[8];
#pragma unroll
  for (int r = 0; r < 8; ++r) { mrow[r] = -3.0e38f; lrow[r] = 0.f; }

  for (int kb = 0; kb < N; kb += ATT_BN) {
    if (kb + ATT_BN < N) {
      __builtin_prefetch(Kb + (size_t)(kb + ATT_BN) * 256, 0, 1);
      __builtin_prefetch(Vb + (size_t)(kb + ATT_BN), 0, 1);
    }
    // S = Q * K^T over Ck=256 (8 k-steps), 4 key tiles of 16
    v8f s[4];
#pragma unroll
    for (int j = 0; j < 4; ++j) s[j] = vzero8();
#pragma unroll
    for (int kc = 0; kc < 8; ++kc) {
      v16bf a = load_fragA(Qw, 256, kc * 32);
#pragma unroll
      for (int j = 0; j < 4; ++j) {
        v16bf bK = load_fragB(Kb + (size_t)(kb + j * 16) * 256, 256, kc * 32);
        s[j] = wmma_bf16(a, bK, s[j]);
      }
    }
#pragma unroll
    for (int j = 0; j < 4; ++j) s[j] = s[j] * 0.0625f;  // Ck^-0.5

    // online softmax: row stats live replicated across the 16 lanes of a half
    float mnew[8], corr[8];
#pragma unroll
    for (int r = 0; r < 8; ++r) {
      float mx = -3.0e38f;
#pragma unroll
      for (int j = 0; j < 4; ++j) mx = fmaxf(mx, s[j][r]);
#pragma unroll
      for (int d = 1; d < 16; d <<= 1) mx = fmaxf(mx, __shfl_xor(mx, d, 32));
      mnew[r] = fmaxf(mrow[r], mx);
      corr[r] = __expf(mrow[r] - mnew[r]);
      mrow[r] = mnew[r];
    }
#pragma unroll
    for (int r = 0; r < 8; ++r) {
      float sum = 0.f;
#pragma unroll
      for (int j = 0; j < 4; ++j) {
        float p = __expf(s[j][r] - mnew[r]);
        sum += p;
        sP[w][r + 8 * hh][j * 16 + r16] = (bf16_t)p;
      }
#pragma unroll
      for (int d = 1; d < 16; d <<= 1) sum += __shfl_xor(sum, d, 32);
      lrow[r] = lrow[r] * corr[r] + sum;
    }
    // rescale running O, then O += P @ V^T (K-dim = 64 keys, 2 k-steps)
#pragma unroll
    for (int t = 0; t < 16; ++t)
#pragma unroll
      for (int r = 0; r < 8; ++r) o[t][r] *= corr[r];
#pragma unroll
    for (int kc = 0; kc < 2; ++kc) {
      v16bf a = load_fragA(&sP[w][0][0], ATT_BN, kc * 32);
#pragma unroll
      for (int t = 0; t < 16; ++t) {
        v16bf bV = load_fragB(Vb + (size_t)(t * 16) * N + kb, N, kc * 32);
        o[t] = wmma_bf16(a, bV, o[t]);
      }
    }
  }

#pragma unroll
  for (int r = 0; r < 8; ++r) lrow[r] = 1.0f / lrow[r];
  bf16_t* Cw = Ct + (size_t)b * N * 256 + (size_t)(q0 + w * 16) * 256;
#pragma unroll
  for (int t = 0; t < 16; ++t)
#pragma unroll
    for (int r = 0; r < 8; ++r)
      Cw[(size_t)(r + 8 * hh) * 256 + t * 16 + r16] = (bf16_t)(o[t][r] * lrow[r]);
}

// ---------------------------------------------------------------------------
// Output GEMM: out[o,n] = wos[o] * (Wo @ Ct^T)[o,n] + wosh[o], f32 out.
// A = Wo rows (v contiguous), B = Ct rows (v contiguous) -> direct global frags.
// ---------------------------------------------------------------------------
__global__ __launch_bounds__(256) void out_kernel(
    const bf16_t* __restrict__ Wo, const bf16_t* __restrict__ Ct,
    const float* __restrict__ wos, const float* __restrict__ wosh,
    float* __restrict__ out, int N) {
  const int b = blockIdx.z;
  const int o0 = blockIdx.y * 128 + (int)(threadIdx.x >> 5) * 16;
  const int n0 = blockIdx.x * 128;
  const int lane = threadIdx.x & 31, r16 = lane & 15, hh = lane >> 4;
  const bf16_t* Cb = Ct + (size_t)b * N * 256;

  v8f acc[8];
#pragma unroll
  for (int t = 0; t < 8; ++t) acc[t] = vzero8();
#pragma unroll
  for (int kc = 0; kc < 8; ++kc) {
    v16bf a = load_fragA(Wo + (size_t)o0 * 256, 256, kc * 32);
#pragma unroll
    for (int nt = 0; nt < 8; ++nt) {
      v16bf bc = load_fragB(Cb + (size_t)(n0 + nt * 16) * 256, 256, kc * 32);
      acc[nt] = wmma_bf16(a, bc, acc[nt]);
    }
  }
#pragma unroll
  for (int r = 0; r < 8; ++r) {
    int oc = o0 + r + 8 * hh;
    float sc = wos[oc], sh = wosh[oc];
    float* op = out + ((size_t)b * 512 + oc) * N + n0;
#pragma unroll
    for (int nt = 0; nt < 8; ++nt) op[nt * 16 + r16] = acc[nt][r] * sc + sh;
  }
}

// ---------------------------------------------------------------------------
// Host launcher
// ---------------------------------------------------------------------------
extern "C" void kernel_launch(void* const* d_in, const int* in_sizes, int n_in,
                              void* d_out, int out_size, void* d_ws, size_t ws_size,
                              hipStream_t stream) {
  (void)in_sizes; (void)n_in; (void)out_size; (void)ws_size;
  const int B = 2, N = 96 * 96;  // per reference

  const float* x      = (const float*)d_in[0];
  const float* fk_w1  = (const float*)d_in[1];
  const float* fk_bn1 = (const float*)d_in[2];
  const float* fk_w2  = (const float*)d_in[3];
  const float* fk_bn2 = (const float*)d_in[4];
  const float* fq_w1  = (const float*)d_in[5];
  const float* fq_bn1 = (const float*)d_in[6];
  const float* fq_w2  = (const float*)d_in[7];
  const float* fq_bn2 = (const float*)d_in[8];
  const float* fv_w   = (const float*)d_in[9];
  const float* wo_w   = (const float*)d_in[10];
  const float* wo_bn  = (const float*)d_in[11];

  // workspace carve-up
  bf16_t* Aq  = (bf16_t*)d_ws;            // 256*512
  bf16_t* Ak  = Aq + 256 * 512;
  bf16_t* WvB = Ak + 256 * 512;
  bf16_t* WoB = WvB + 256 * 512;
  float*  bq  = (float*)(WoB + 512 * 256);
  float*  bk  = bq + 256;
  float*  P   = bk + 256;                 // 3072 floats of BN params
  bf16_t* Qt  = (bf16_t*)(P + 3072);      // [B, N, 256]
  bf16_t* Kt  = Qt + (size_t)B * N * 256; // [B, N, 256]
  bf16_t* Vb  = Kt + (size_t)B * N * 256; // [B, 256, N]
  bf16_t* Ctb = Vb + (size_t)B * 256 * N; // [B, N, 256]

  bnparams_kernel<<<6, 256, 0, stream>>>(fk_bn1, fk_bn2, fq_bn1, fq_bn2, wo_bn, P);
  fold_kernel<<<512, 256, 0, stream>>>(fq_w1, fq_w2, P + 1024, P + 1280,
                                       P + 1536, P + 1792, Aq, bq);
  fold_kernel<<<512, 256, 0, stream>>>(fk_w1, fk_w2, P + 0, P + 256,
                                       P + 512, P + 768, Ak, bk);
  cvt_kernel<<<(256 * 512 + 255) / 256, 256, 0, stream>>>(fv_w, WvB, 256 * 512);
  cvt_kernel<<<(512 * 256 + 255) / 256, 256, 0, stream>>>(wo_w, WoB, 512 * 256);

  proj_kernel<<<dim3(N / 128, B, 3), 256, 0, stream>>>(x, Aq, Ak, WvB, bq, bk,
                                                       Qt, Kt, Vb, N);
  attn_kernel<<<dim3(N / 128, B), 256, 0, stream>>>(Qt, Kt, Vb, Ctb, N);
  out_kernel<<<dim3(N / 128, 512 / 128, B), 256, 0, stream>>>(
      WoB, Ctb, P + 2048, P + 2560, (float*)d_out, N);
}
